// DepGCN_15582141350765
// MI455X (gfx1250) — matrix operand order repaired and checked by reading, not compiled, not run on hardware
//
#include <hip/hip_runtime.h>
#include <hip/hip_bf16.h>

// Reference collapses algebraically to: relu(text + (emb @ fc_w + fc_b)[dep_labels])
//  - attention score is constant over the softmax axis -> cancels in softmax
//  - softmax rows sum to 1 and multiply a j-constant value -> identity
//  - e @ fc_w = (emb @ fc_w)[dep_labels] since labels index a 50-row table
//
// Kernel 1: table[50,512] = emb[50,64] @ fc_w[64,512] + fc_b   (WMMA f32<-f16)
// Kernel 2: out[b,s,f] = relu(text[b,s,f] + table[dep_labels[b,s]][f])  (mem-bound)

#define DEP_NUM 50
#define DEP_DIM 64
#define OUT_F   512
#define NROWS   (4 * 2048)          // B*S
#define OUT_ELEMS (NROWS * OUT_F)   // 4,194,304 floats

typedef __attribute__((ext_vector_type(16))) _Float16 v16h;
typedef __attribute__((ext_vector_type(8)))  float    v8f;

// ---------------------------------------------------------------------------
// Kernel 1: 64x64x512 table GEMM via V_WMMA_F32_16X16X32_F16.
// One wave (32 lanes) computes one 16x16 tile of D; K=64 -> two k-steps of 32.
// Grid: 4 m-tiles * 32 n-tiles = 128 waves.
// ---------------------------------------------------------------------------
__global__ __launch_bounds__(32) void depgcn_table_wmma(
    const float* __restrict__ emb,    // [50,64]
    const float* __restrict__ fc_w,   // [64,512]
    const float* __restrict__ fc_b,   // [512]
    float* __restrict__ table)        // [50,512] (ws)
{
  const int tile = blockIdx.x;        // 0..127
  const int mt   = tile >> 5;         // 0..3
  const int nt   = tile & 31;         // 0..31
  const int lane = threadIdx.x;       // 0..31
  const int lo   = lane & 15;
  const bool hi  = lane >= 16;

  const int m = mt * 16 + lo;         // A row (shared by both lane halves)
  const int n = nt * 16 + lo;         // B/D column

  v8f c = {};

  #pragma unroll
  for (int k0 = 0; k0 < DEP_DIM; k0 += 32) {
    // ---- A fragment: 16x32 f16, contiguous K (same order as B) ----
    // lanes 0-15 : h[0..15] = K 0..15
    // lanes 16-31: h[0..15] = K 16..31
    v16h a;
    const int koffA = hi ? 16 : 0;
    if (m < DEP_NUM) {
      const float* arow = emb + m * DEP_DIM + k0;
      #pragma unroll
      for (int j = 0; j < 8; ++j) {
        a[j]     = (_Float16)arow[koffA + j];
        a[8 + j] = (_Float16)arow[koffA + 8 + j];
      }
    } else {
      #pragma unroll
      for (int j = 0; j < 16; ++j) a[j] = (_Float16)0.0f;
    }

    // ---- B fragment: 32x16 f16, lane = N, lane-half splits K ----
    // lanes 0-15 : h[j] = fc_w[k0 +      j][n]   (K 0..15)
    // lanes 16-31: h[j] = fc_w[k0 + 16 + j][n]   (K 16..31)
    v16h b;
    const int koffB = hi ? 16 : 0;
    #pragma unroll
    for (int j = 0; j < 16; ++j)
      b[j] = (_Float16)fc_w[(k0 + koffB + j) * OUT_F + n];

    // EXEC is all-1s here (row guard reconverged above).
    c = __builtin_amdgcn_wmma_f32_16x16x32_f16(
        /*neg_a=*/false, a, /*neg_b=*/false, b,
        /*c_mod=*/(short)0, c, /*reuse_a=*/false, /*reuse_b=*/false);
  }

  // ---- D layout: VGPR r -> M = r (lanes 0-15) / M = 8+r (lanes 16-31), N = lane%16
  const float bias = fc_b[n];
  #pragma unroll
  for (int r = 0; r < 8; ++r) {
    const int mm = mt * 16 + (hi ? 8 + r : r);
    if (mm < DEP_NUM) table[mm * OUT_F + n] = c[r] + bias;
  }
}

// ---------------------------------------------------------------------------
// Kernel 2: out = relu(text + table[label])  -- pure streaming, float4 path.
// 32 MB total traffic @ 23.3 TB/s  ->  ~1.4 us floor.
// ---------------------------------------------------------------------------
__global__ __launch_bounds__(256) void depgcn_fuse(
    const float4* __restrict__ text4,   // [NROWS*128] float4
    const int*    __restrict__ labels,  // [NROWS]
    const float*  __restrict__ table,   // [50,512]
    float4*       __restrict__ out4)    // [NROWS*128] float4
{
  const int idx = blockIdx.x * 256 + threadIdx.x;   // < NROWS*128
  const int row = idx >> 7;                         // 128 float4 per row
  const int col = idx & 127;

  const int lab = labels[row];                      // uniform per 128 lanes, L0/L2 hit
  const float4 t = text4[idx];
  const float4 w = ((const float4*)(table + lab * OUT_F))[col];

  float4 o;
  o.x = fmaxf(t.x + w.x, 0.0f);
  o.y = fmaxf(t.y + w.y, 0.0f);
  o.z = fmaxf(t.z + w.z, 0.0f);
  o.w = fmaxf(t.w + w.w, 0.0f);
  out4[idx] = o;
}

extern "C" void kernel_launch(void* const* d_in, const int* in_sizes, int n_in,
                              void* d_out, int out_size, void* d_ws, size_t ws_size,
                              hipStream_t stream) {
  // setup_inputs order: text, dep_mat, dep_labels, emb, attn_w, attn_b, fc_w, fc_b
  const float* text   = (const float*)d_in[0];
  // d_in[1] (dep_mat), d_in[4] (attn_w), d_in[5] (attn_b) are provably unused.
  const int*   labels = (const int*)  d_in[2];
  const float* emb    = (const float*)d_in[3];
  const float* fc_w   = (const float*)d_in[6];
  const float* fc_b   = (const float*)d_in[7];
  float* out   = (float*)d_out;
  float* table = (float*)d_ws;        // 50*512*4 = 100 KB scratch

  depgcn_table_wmma<<<128, 32, 0, stream>>>(emb, fc_w, fc_b, table);

  const int n4 = OUT_ELEMS / 4;       // 1,048,576 float4
  depgcn_fuse<<<n4 / 256, 256, 0, stream>>>(
      (const float4*)text, labels, table, (float4*)out);
}
